// ALBEF_32959579029881
// MI455X (gfx1250) — matrix operand order repaired and checked by reading, not compiled, not run on hardware
//
#include <hip/hip_runtime.h>
#include <hip/hip_bf16.h>

// MI455X (gfx1250, wave32).
// Big GEMMs -> v_wmma_f32_16x16x32_f16 (f16 in, f32 accum), 64x256 block tile,
// 8 waves x (32x64) C-tiles, LDS double-buffered, async global->LDS loads
// (ASYNCcnt) when the toolchain exposes them, else register-staged pipeline.

typedef _Float16 h16;
typedef __attribute__((ext_vector_type(16))) _Float16 v16h;
typedef __attribute__((ext_vector_type(8)))  _Float16 v8h;
typedef __attribute__((ext_vector_type(8)))  float    v8f;
typedef __attribute__((ext_vector_type(4)))  int      v4i;

#if defined(__has_builtin)
#  if __has_builtin(__builtin_amdgcn_global_load_async_to_lds_b128)
#    define ASYNC_LDS 1
#  endif
#endif
#ifndef ASYNC_LDS
#  define ASYNC_LDS 0
#endif

#if ASYNC_LDS
typedef __attribute__((address_space(1))) v4i gv4i;   // global
typedef __attribute__((address_space(3))) v4i lv4i;   // LDS
__device__ __forceinline__ void cp16_async(h16* lds, const h16* g)
{
    __builtin_amdgcn_global_load_async_to_lds_b128((gv4i*)g, (lv4i*)lds, 0, 0);
}
__device__ __forceinline__ void wait_async0()
{
#  if __has_builtin(__builtin_amdgcn_s_wait_asynccnt)
    __builtin_amdgcn_s_wait_asynccnt(0);
#  else
    asm volatile("s_wait_asynccnt 0x0" ::: "memory");
#  endif
}
#endif

#define SHUF16(lo, hi) __builtin_shufflevector(lo, hi, 0,1,2,3,4,5,6,7,8,9,10,11,12,13,14,15)

// ---------------------------------------------------------------------------
// C[M,N] = act(A[M,K] @ B[K,N] + bias); B pre-transposed (Bt[N][K], ldb=K-dim).
// Grid: x = N/256, y = M/64, z = batch. Block 256 thr = 8 waves.
// Wave w: M strip (w&1)*32, N strip (w>>1)*64 -> 2x4 16x16 tiles, 8 WMMA/step.
// ---------------------------------------------------------------------------
__global__ __launch_bounds__(256)
void gemm_f16_wmma(const h16* __restrict__ A, int lda, long long strideA,
                   const h16* __restrict__ Bt, int ldb, long long strideB,
                   float* __restrict__ Cf32, int ldc32,
                   h16* __restrict__ Cf16, int ldc16, long long strideC,
                   const float* __restrict__ bias, int act,
                   int Mstore, int K)
{
    __shared__ __align__(16) h16 As[2][64][40];    // 64 M x 32 K (+8 pad)
    __shared__ __align__(16) h16 Bs[2][256][40];   // 256 N x 32 K (+8 pad)

    const int zb = blockIdx.z;
    A  += (long long)zb * strideA;
    Bt += (long long)zb * strideB;
    const long long coff = (long long)zb * strideC;

    const int m0 = blockIdx.y * 64;
    const int n0 = blockIdx.x * 256;
    const int tid  = threadIdx.x;
    const int w    = tid >> 5;
    const int lane = tid & 31;
    const int lr   = lane & 15;
    const int sel  = lane >> 4;
    const int mw   = (w & 1) * 32;     // wave M offset in block
    const int nw   = (w >> 1) * 64;    // wave N offset in block

    const int ldrow = tid >> 2;          // 0..63
    const int ldcol = (tid & 3) * 8;     // 0,8,16,24

    v8f acc[2][4] = {};

#if ASYNC_LDS
    auto load_tiles = [&](int k0, int buf) {
        cp16_async(&As[buf][ldrow][ldcol],
                   A + (long long)(m0 + ldrow) * lda + k0 + ldcol);
        #pragma unroll
        for (int q = 0; q < 4; ++q)
            cp16_async(&Bs[buf][ldrow + 64 * q][ldcol],
                       Bt + (long long)(n0 + ldrow + 64 * q) * ldb + k0 + ldcol);
    };
#else
    uint4 ra, rb0, rb1, rb2, rb3;
    auto fetch = [&](int k0) {
        ra  = *(const uint4*)(A  + (long long)(m0 + ldrow) * lda + k0 + ldcol);
        rb0 = *(const uint4*)(Bt + (long long)(n0 + ldrow)       * ldb + k0 + ldcol);
        rb1 = *(const uint4*)(Bt + (long long)(n0 + ldrow +  64) * ldb + k0 + ldcol);
        rb2 = *(const uint4*)(Bt + (long long)(n0 + ldrow + 128) * ldb + k0 + ldcol);
        rb3 = *(const uint4*)(Bt + (long long)(n0 + ldrow + 192) * ldb + k0 + ldcol);
    };
    auto stage = [&](int buf) {
        *(uint4*)(&As[buf][ldrow][ldcol])       = ra;
        *(uint4*)(&Bs[buf][ldrow][ldcol])       = rb0;
        *(uint4*)(&Bs[buf][ldrow +  64][ldcol]) = rb1;
        *(uint4*)(&Bs[buf][ldrow + 128][ldcol]) = rb2;
        *(uint4*)(&Bs[buf][ldrow + 192][ldcol]) = rb3;
    };
#endif

    auto compute = [&](int buf) {
        v16h af[2];
        #pragma unroll
        for (int mi = 0; mi < 2; ++mi) {
            const h16* ap = &As[buf][mw + mi * 16 + lr][0];
            v8h lo = *(const v8h*)(ap + 8 * sel);
            v8h hi = *(const v8h*)(ap + 16 + 8 * sel);
            af[mi] = SHUF16(lo, hi);
        }
        #pragma unroll
        for (int ni = 0; ni < 4; ++ni) {
            const h16* bp = &Bs[buf][nw + ni * 16 + lr][0];
            v8h bl = *(const v8h*)(bp + 8 * sel);
            v8h bh = *(const v8h*)(bp + 16 + 8 * sel);
            v16h bf = SHUF16(bl, bh);
            acc[0][ni] = __builtin_amdgcn_wmma_f32_16x16x32_f16(false, af[0], false, bf,
                                                               (short)0, acc[0][ni], false, false);
            acc[1][ni] = __builtin_amdgcn_wmma_f32_16x16x32_f16(false, af[1], false, bf,
                                                               (short)0, acc[1][ni], false, false);
        }
    };

    // software-pipelined, LDS double-buffered main loop
#if ASYNC_LDS
    load_tiles(0, 0);
    wait_async0();
    __syncthreads();
    int cur = 0;
    for (int k0 = 0; k0 < K; k0 += 32) {
        const bool has_next = (k0 + 32) < K;
        if (has_next) load_tiles(k0 + 32, cur ^ 1);   // async loads overlap WMMAs
        compute(cur);
        if (has_next) wait_async0();
        __syncthreads();
        cur ^= 1;
    }
#else
    fetch(0);
    stage(0);
    __syncthreads();
    int cur = 0;
    for (int k0 = 0; k0 < K; k0 += 32) {
        const bool has_next = (k0 + 32) < K;
        if (has_next) fetch(k0 + 32);       // global loads overlap the WMMAs
        compute(cur);
        if (has_next) stage(cur ^ 1);       // load-wait deferred to here
        __syncthreads();
        cur ^= 1;
    }
#endif

    // C layout: VGPR r -> M = r (+8 for lanes 16-31), N = lane&15
    #pragma unroll
    for (int mi = 0; mi < 2; ++mi) {
        #pragma unroll
        for (int r = 0; r < 8; ++r) {
            const int mg = m0 + mw + mi * 16 + r + 8 * sel;
            if (mg >= Mstore) continue;
            #pragma unroll
            for (int ni = 0; ni < 4; ++ni) {
                const int ng = n0 + nw + ni * 16 + lr;
                float v = acc[mi][ni][r];
                if (bias) v += bias[ng];
                if      (act == 1) v = v > 0.f ? v : (__expf(v) - 1.f);   // ELU
                else if (act == 2) v = v > 0.f ? v : 0.f;                 // ReLU
                else if (act == 3) v = 1.f / (1.f + __expf(-v));          // sigmoid
                if (Cf32) Cf32[coff + (long long)mg * ldc32 + ng] = v;
                if (Cf16) Cf16[coff + (long long)mg * ldc16 + ng] = (h16)v;
            }
        }
    }
}

// ---------------------------------------------------------------------------
// LDS-tiled transpose + fp16 convert with zero padding:
// out[r][c] = (r < inCols && c < inRows) ? in[c][r] : 0
// ---------------------------------------------------------------------------
template <typename TIn>
__global__ void transpose_to_f16(const TIn* __restrict__ in, long long inStride,
                                 int inRows, int inCols, int ldi,
                                 h16* __restrict__ out, long long outStride,
                                 int outRows, int outCols, int ldo)
{
    __shared__ float tile[32][33];
    in  += (long long)blockIdx.z * inStride;
    out += (long long)blockIdx.z * outStride;
    const int c0 = blockIdx.x * 32;
    const int r0 = blockIdx.y * 32;
    const int tx = threadIdx.x, ty = threadIdx.y;   // (32, 8)
    for (int i = ty; i < 32; i += 8) {
        const int rr = r0 + i, cc = c0 + tx;
        tile[i][tx] = (rr < inRows && cc < inCols)
                      ? (float)in[(long long)rr * ldi + cc] : 0.f;
    }
    __syncthreads();
    for (int i = ty; i < 32; i += 8) {
        const int orow = c0 + i, ocol = r0 + tx;
        if (orow < outRows && ocol < outCols)
            out[(long long)orow * ldo + ocol] = (h16)tile[tx][i];
    }
}

// Pack image[:,1:,:] then text[:,1:,:] into one flat f16 A matrix [40896,768]
__global__ void pack_x(const float* __restrict__ img, const float* __restrict__ txt,
                       h16* __restrict__ X16)
{
    const long long idx = (long long)blockIdx.x * 256 + threadIdx.x;
    const long long m = idx / 768;
    const int d = (int)(idx % 768);
    float v;
    if (m < 36864) {
        const long long b = m / 576, n = m % 576;
        v = img[(b * 577 + n + 1) * 768 + d];
    } else {
        const long long r = m - 36864, b = r / 63, n = r % 63;
        v = txt[(b * 64 + n + 1) * 768 + d];
    }
    X16[idx] = (h16)v;
}

// per-row dual dot: o0[row]=h.a0, o1[row]=h.a1 (el / er terms)
__global__ __launch_bounds__(256)
void row_dot2(const h16* __restrict__ H, int ld,
              const float* __restrict__ a0, const float* __restrict__ a1,
              float* __restrict__ o0, float* __restrict__ o1, int K)
{
    const long long row = blockIdx.x;
    const h16* h = H + row * ld;
    float s0 = 0.f, s1 = 0.f;
    for (int k = threadIdx.x; k < K; k += 256) {
        const float v = (float)h[k];
        s0 += v * a0[k]; s1 += v * a1[k];
    }
    __shared__ float r0[256], r1[256];
    r0[threadIdx.x] = s0; r1[threadIdx.x] = s1; __syncthreads();
    for (int s = 128; s > 0; s >>= 1) {
        if (threadIdx.x < s) { r0[threadIdx.x] += r0[threadIdx.x + s];
                               r1[threadIdx.x] += r1[threadIdx.x + s]; }
        __syncthreads();
    }
    if (threadIdx.x == 0) { o0[row] = r0[0]; o1[row] = r1[0]; }
}

// alphaT[b][j][i] = softmax_i( leaky_relu(el[i]+er[j]) , i != j ), f16, padded
__global__ __launch_bounds__(256)
void build_alphaT(const float* __restrict__ el, const float* __restrict__ er,
                  h16* __restrict__ alphaT, int N, int Np)
{
    const int b = blockIdx.y, j = blockIdx.x, tid = threadIdx.x;
    h16* outRow = alphaT + ((long long)b * Np + j) * Np;
    if (j >= N) { for (int i = tid; i < Np; i += 256) outRow[i] = (h16)0.f; return; }
    const float* elb = el + (long long)b * N;
    const float erj = er[(long long)b * N + j];
    __shared__ float red[256];

    float mx = -3.0e38f;
    for (int i = tid; i < N; i += 256) {
        if (i == j) continue;
        float t = elb[i] + erj; t = t > 0.f ? t : 0.2f * t;
        mx = fmaxf(mx, t);
    }
    red[tid] = mx; __syncthreads();
    for (int s = 128; s > 0; s >>= 1) { if (tid < s) red[tid] = fmaxf(red[tid], red[tid + s]); __syncthreads(); }
    mx = red[0]; __syncthreads();

    float sm = 0.f;
    for (int i = tid; i < N; i += 256) {
        if (i == j) continue;
        float t = elb[i] + erj; t = t > 0.f ? t : 0.2f * t;
        sm += __expf(t - mx);
    }
    red[tid] = sm; __syncthreads();
    for (int s = 128; s > 0; s >>= 1) { if (tid < s) red[tid] += red[tid + s]; __syncthreads(); }
    const float inv = 1.f / red[0];

    for (int i = tid; i < Np; i += 256) {
        float v = 0.f;
        if (i < N && i != j) {
            float t = elb[i] + erj; t = t > 0.f ? t : 0.2f * t;
            v = __expf(t - mx) * inv;
        }
        outRow[i] = (h16)v;
    }
}

// Stage-2 GAT, destination node 0 only
__global__ __launch_bounds__(256)
void gat_node0(const float* __restrict__ elf, const float* __restrict__ erf,
               const h16* __restrict__ Hf, const float* __restrict__ b_g,
               float* __restrict__ gat0elu)
{
    const int b = blockIdx.x, tid = threadIdx.x;
    const long long base = (long long)b * 639;
    const float er0 = erf[base];
    __shared__ float wgt[640];
    __shared__ float red[256];

    float mx = -3.0e38f;
    for (int i = 1 + tid; i < 639; i += 256) {
        float t = elf[base + i] + er0; t = t > 0.f ? t : 0.2f * t;
        mx = fmaxf(mx, t);
    }
    red[tid] = mx; __syncthreads();
    for (int s = 128; s > 0; s >>= 1) { if (tid < s) red[tid] = fmaxf(red[tid], red[tid + s]); __syncthreads(); }
    mx = red[0]; __syncthreads();

    float sm = 0.f;
    for (int i = 1 + tid; i < 639; i += 256) {
        float t = elf[base + i] + er0; t = t > 0.f ? t : 0.2f * t;
        const float e = __expf(t - mx);
        wgt[i] = e; sm += e;
    }
    red[tid] = sm; __syncthreads();
    for (int s = 128; s > 0; s >>= 1) { if (tid < s) red[tid] += red[tid + s]; __syncthreads(); }
    const float inv = 1.f / red[0];
    __syncthreads();

    float acc = 0.f;
    for (int i = 1; i < 639; ++i)
        acc += wgt[i] * (float)Hf[(base + i) * 256 + tid];
    acc = acc * inv + b_g[tid];
    acc = acc > 0.f ? acc : (__expf(acc) - 1.f);
    gat0elu[b * 256 + tid] = acc;
}

__global__ void ge0_head(const float* __restrict__ gat0elu, const float* __restrict__ W_c,
                         const float* __restrict__ b_c, h16* __restrict__ ge0r16)
{
    const int b = blockIdx.x, c = threadIdx.x;      // 64 x 128
    float s = b_c[c];
    for (int k = 0; k < 256; ++k) s += gat0elu[b * 256 + k] * W_c[k * 128 + c];
    s = s > 0.f ? s : 0.f;
    ge0r16[b * 128 + c] = (h16)s;
}

__global__ void mul_kernel(const float* __restrict__ txt, float* __restrict__ mul32,
                           h16* __restrict__ fusion16)
{
    const int idx = blockIdx.x * 256 + threadIdx.x;   // 64*768
    const int b = idx / 768, d = idx % 768;
    float v = txt[(long long)b * 64 * 768 + d];
    v = v > 0.f ? v : 0.f;
    mul32[idx] = v;
    fusion16[(long long)b * 1536 + d] = (h16)v;
}

__global__ void select_kernel(const float* __restrict__ soft32, const float* __restrict__ gum,
                              const float* __restrict__ mul32, const float* __restrict__ graphv32,
                              h16* __restrict__ pre16)
{
    const int idx = blockIdx.x * 256 + threadIdx.x;   // 64*768
    const int b = idx / 768, d = idx % 768;
    const float s0 = soft32[(long long)b * 1536 + d];
    const float s1 = soft32[(long long)b * 1536 + 768 + d];
    const float g0 = gum[(long long)b * 1536 + d];
    const float g1 = gum[(long long)b * 1536 + 768 + d];
    const float pre = (s0 + g0 >= s1 + g1) ? mul32[idx] : graphv32[idx];
    pre16[idx] = (h16)pre;
}

__global__ void final_pred(const h16* __restrict__ c1out16, const float* __restrict__ W_c2,
                           const float* __restrict__ b_c2, float* __restrict__ out)
{
    const int t = threadIdx.x;              // 128 = 64*2
    const int b = t >> 1, c = t & 1;
    float s = b_c2[c];
    for (int k = 0; k < 768; ++k) s += (float)c1out16[b * 768 + k] * W_c2[k * 2 + c];
    out[b * 2 + c] = s;
}

// ---------------------------------------------------------------------------
extern "C" void kernel_launch(void* const* d_in, const int* in_sizes, int n_in,
                              void* d_out, int out_size, void* d_ws, size_t ws_size,
                              hipStream_t stream)
{
    const float* img  = (const float*)d_in[0];
    const float* txt  = (const float*)d_in[1];
    const float* gum  = (const float*)d_in[2];
    const float* W_p  = (const float*)d_in[3];
    const float* b_p  = (const float*)d_in[4];
    const float* al_p = (const float*)d_in[5];
    const float* ar_p = (const float*)d_in[6];
    const float* W_g  = (const float*)d_in[7];
    const float* b_g  = (const float*)d_in[8];
    const float* al_g = (const float*)d_in[9];
    const float* ar_g = (const float*)d_in[10];
    const float* W_c  = (const float*)d_in[11];
    const float* b_c  = (const float*)d_in[12];
    const float* W_hp = (const float*)d_in[13];
    const float* b_hp = (const float*)d_in[14];
    const float* W_hm = (const float*)d_in[15];
    const float* b_hm = (const float*)d_in[16];
    const float* W_hg = (const float*)d_in[17];
    const float* b_hg = (const float*)d_in[18];
    const float* W_c1 = (const float*)d_in[19];
    const float* b_c1 = (const float*)d_in[20];
    const float* W_c2 = (const float*)d_in[21];
    const float* b_c2 = (const float*)d_in[22];
    float* out = (float*)d_out;

    const long long Mi = 64LL * 576;     // 36864 image rows
    const long long Mt = 64LL * 63;      // 4032 text rows
    const long long Mall = Mi + Mt;      // 40896 (multiple of 64)

    char* ws = (char*)d_ws;
    size_t off = 0;
    auto alloc = [&](size_t bytes) -> void* {
        off = (off + 255) & ~(size_t)255;
        void* p = ws + off; off += bytes; return p;
    };

    h16*   X16   = (h16*)  alloc(Mall * 768 * sizeof(h16));       // aliased by proj16
    h16*   WpT   = (h16*)  alloc(768LL * 768 * sizeof(h16));
    h16*   H16   = (h16*)  alloc(Mall * 768 * sizeof(h16));       // aliased by Hf16
    float* el_p  = (float*)alloc(Mall * sizeof(float));
    float* er_p  = (float*)alloc(Mall * sizeof(float));
    h16*   aTi   = (h16*)  alloc(64LL * 576 * 576 * sizeof(h16));
    h16*   aTt   = (h16*)  alloc(64LL * 64 * 64 * sizeof(h16));
    h16*   hTi   = (h16*)  alloc(64LL * 768 * 576 * sizeof(h16));
    h16*   hTt   = (h16*)  alloc(64LL * 768 * 64 * sizeof(h16));
    h16*   WgT   = (h16*)  alloc(256LL * 768 * sizeof(h16));
    float* elf   = (float*)alloc(Mall * sizeof(float));
    float* erf   = (float*)alloc(Mall * sizeof(float));
    float* g0e   = (float*)alloc(64LL * 256 * sizeof(float));
    h16*   ge0r  = (h16*)  alloc(64LL * 128 * sizeof(h16));
    h16*   WhpT  = (h16*)  alloc(768LL * 128 * sizeof(h16));
    float* grv   = (float*)alloc(64LL * 768 * sizeof(float));
    h16*   fus   = (h16*)  alloc(64LL * 1536 * sizeof(h16));
    h16*   WhmT  = (h16*)  alloc(768LL * 1536 * sizeof(h16));
    h16*   WhgT  = (h16*)  alloc(768LL * 1536 * sizeof(h16));
    float* soft  = (float*)alloc(64LL * 1536 * sizeof(float));
    float* mul32 = (float*)alloc(64LL * 768 * sizeof(float));
    h16*   pre16 = (h16*)  alloc(64LL * 768 * sizeof(h16));
    h16*   Wc1T  = (h16*)  alloc(768LL * 768 * sizeof(h16));
    h16*   c1o   = (h16*)  alloc(64LL * 768 * sizeof(h16));
    h16*   proj16 = X16;                  // X16 dead after G1
    h16*   Hf16   = H16;                  // H16 dead after hT transposes

    const dim3 B256(256), Btr(32, 8);

    // 1) pack A, transpose-convert all weights
    pack_x<<<(unsigned)(Mall * 768 / 256), B256, 0, stream>>>(img, txt, X16);
    transpose_to_f16<float><<<dim3(24, 24, 1), Btr, 0, stream>>>(W_p, 0, 768, 768, 768, WpT, 0, 768, 768, 768);
    transpose_to_f16<float><<<dim3(8, 24, 1), Btr, 0, stream>>>(W_g, 0, 768, 256, 256, WgT, 0, 256, 768, 768);
    transpose_to_f16<float><<<dim3(24, 4, 1), Btr, 0, stream>>>(W_hp, 0, 128, 768, 768, WhpT, 0, 768, 128, 128);
    transpose_to_f16<float><<<dim3(24, 48, 1), Btr, 0, stream>>>(W_hm, 0, 1536, 768, 768, WhmT, 0, 768, 1536, 1536);
    transpose_to_f16<float><<<dim3(24, 48, 1), Btr, 0, stream>>>(W_hg, 0, 1536, 768, 768, WhgT, 0, 768, 1536, 1536);
    transpose_to_f16<float><<<dim3(24, 24, 1), Btr, 0, stream>>>(W_c1, 0, 768, 768, 768, Wc1T, 0, 768, 768, 768);

    // 2) G1: H = X @ W_p   [40896,768]x[768,768]
    gemm_f16_wmma<<<dim3(3, 639, 1), B256, 0, stream>>>(
        X16, 768, 0, WpT, 768, 0, nullptr, 0, H16, 768, 0, nullptr, 0, (int)Mall, 768);

    // 3) el/er per row, softmax rows (alphaT), per-sample h^T
    row_dot2<<<(unsigned)Mall, B256, 0, stream>>>(H16, 768, al_p, ar_p, el_p, er_p, 768);
    build_alphaT<<<dim3(576, 64), B256, 0, stream>>>(el_p, er_p, aTi, 576, 576);
    build_alphaT<<<dim3(64, 64), B256, 0, stream>>>(el_p + Mi, er_p + Mi, aTt, 63, 64);
    transpose_to_f16<h16><<<dim3(24, 18, 64), Btr, 0, stream>>>(
        H16, 576LL * 768, 576, 768, 768, hTi, 768LL * 576, 768, 576, 576);
    transpose_to_f16<h16><<<dim3(24, 2, 64), Btr, 0, stream>>>(
        H16 + Mi * 768, 63LL * 768, 63, 768, 768, hTt, 768LL * 64, 768, 64, 64);

    // 4) stage-1 aggregation: proj = elu(alphaT @ h + b_p), batched per sample
    gemm_f16_wmma<<<dim3(3, 9, 64), B256, 0, stream>>>(
        aTi, 576, 576LL * 576, hTi, 576, 768LL * 576,
        nullptr, 0, proj16, 768, 576LL * 768, b_p, 1, 576, 576);
    gemm_f16_wmma<<<dim3(3, 1, 64), B256, 0, stream>>>(
        aTt, 64, 64LL * 64, hTt, 64, 768LL * 64,
        nullptr, 0, proj16 + Mi * 768, 768, 63LL * 768, b_p, 1, 63, 64);

    // 5) G3: h_f = proj @ W_g   [40896,768]x[768,256]
    gemm_f16_wmma<<<dim3(1, 639, 1), B256, 0, stream>>>(
        proj16, 768, 0, WgT, 768, 0, nullptr, 0, Hf16, 256, 0, nullptr, 0, (int)Mall, 768);

    // 6) stage-2 el/er; node-0 softmax-aggregate; classifier column
    row_dot2<<<(unsigned)Mall, B256, 0, stream>>>(Hf16, 256, al_g, ar_g, elf, erf, 256);
    gat_node0<<<64, B256, 0, stream>>>(elf, erf, Hf16, b_g, g0e);
    ge0_head<<<64, 128, 0, stream>>>(g0e, W_c, b_c, ge0r);

    // 7) fusion head
    mul_kernel<<<192, B256, 0, stream>>>(txt, mul32, fus);
    gemm_f16_wmma<<<dim3(3, 1, 1), B256, 0, stream>>>(
        ge0r, 128, 0, WhpT, 128, 0, grv, 768, fus + 768, 1536, 0, b_hp, 0, 64, 128);
    gemm_f16_wmma<<<dim3(3, 1, 1), B256, 0, stream>>>(
        fus, 1536, 0, WhmT, 1536, 0, soft, 1536, nullptr, 0, 0, b_hm, 3, 64, 1536);
    gemm_f16_wmma<<<dim3(3, 1, 1), B256, 0, stream>>>(
        fus, 1536, 0, WhgT, 1536, 0, soft + 768, 1536, nullptr, 0, 0, b_hg, 3, 64, 1536);
    select_kernel<<<192, B256, 0, stream>>>(soft, gum, mul32, grv, pre16);

    // 8) cls head
    gemm_f16_wmma<<<dim3(3, 1, 1), B256, 0, stream>>>(
        pre16, 768, 0, Wc1T, 768, 0, nullptr, 0, c1o, 768, 0, b_c1, 2, 64, 768);
    final_pred<<<1, 128, 0, stream>>>(c1o, W_c2, b_c2, out);
}
